// NemotronTopkRouter_60782377173310
// MI455X (gfx1250) — matrix operand order repaired and checked by reading, not compiled, not run on hardware
//
#include <hip/hip_runtime.h>
#include <hip/hip_bf16.h>
#include <math.h>

// ---------------------------------------------------------------------------
// Nemotron/DeepSeek MoE top-k router for MI455X (gfx1250, wave32, WMMA).
//
// logits = hidden(32768x2048 f32) x weight^T(64x2048 f32)   [memory bound:
// 256 MB of hidden @ 23.3 TB/s ~= 11us floor; needs ~745 TF/s to saturate,
// so we use bf16 WMMA with a 2-term hi/lo split for ~fp32 accuracy].
// Then sigmoid + grouped top-k routing per token.
// ---------------------------------------------------------------------------

typedef __bf16 v16bf __attribute__((ext_vector_type(16)));
typedef float  v8f   __attribute__((ext_vector_type(8)));
typedef float  v4f   __attribute__((ext_vector_type(4)));

#define N_TOKENS    32768
#define DIM         2048
#define N_EXPERTS   64
#define TOP_K       8
#define N_GROUP     8
#define TOPK_GROUP  4
#define ROUTE_SCALE 2.5f

#define WAVES_PER_BLOCK 8
#define TOKENS_PER_WAVE 16
#define BLOCK_TOKENS    (WAVES_PER_BLOCK * TOKENS_PER_WAVE)  // 128
#define LDS_STRIDE      68   // 64 + pad: bank = (68*row + e) % 64 conflict-free

// ---------------------------------------------------------------------------
// Pre-kernel: split fp32 weights into bf16 hi/lo planes (512 KB total in ws).
// hi = rne(x) to bf16 ; lo = rne(x - float(hi)).  x ~= hi + lo to ~16 mantissa
// bits, so  a.b ~= ah.bh + ah.bl + al.bh  recovers near-fp32 dot products.
// ---------------------------------------------------------------------------
__global__ void split_weights_bf16(const float* __restrict__ w,
                                   __bf16* __restrict__ whi,
                                   __bf16* __restrict__ wlo) {
    int i = blockIdx.x * blockDim.x + threadIdx.x;
    if (i < N_EXPERTS * DIM) {
        float x = w[i];
        __bf16 h = (__bf16)x;
        whi[i] = h;
        wlo[i] = (__bf16)(x - (float)h);
    }
}

// ---------------------------------------------------------------------------
// Main fused kernel: GEMM (bf16x3 WMMA) + sigmoid + grouped top-k routing.
// Each wave: 16 tokens x 64 experts (4 N-tiles), K-loop in steps of 32.
// ---------------------------------------------------------------------------
__global__ __launch_bounds__(256) void router_fused(
    const float*  __restrict__ A,      // [N_TOKENS][DIM]
    const __bf16* __restrict__ Whi,    // [N_EXPERTS][DIM]
    const __bf16* __restrict__ Wlo,    // [N_EXPERTS][DIM]
    const float*  __restrict__ bias,   // [N_EXPERTS]
    float*        __restrict__ out)    // [N_TOKENS*8] idx ++ [N_TOKENS*8] w
{
    __shared__ float smem[WAVES_PER_BLOCK * TOKENS_PER_WAVE * LDS_STRIDE];

    const int tid  = threadIdx.x;
    const int wave = tid >> 5;          // 0..7
    const int ln   = tid & 31;          // lane in wave32
    const int m    = ln & 15;           // M row (A) / N col (B) within tile
    const int kh   = ln >> 4;           // K-half select per WMMA 16-bit layout
    const int tokenBase = blockIdx.x * BLOCK_TOKENS + wave * TOKENS_PER_WAVE;

    const float* arow = A + (size_t)(tokenBase + m) * DIM;

    v8f acc[4];
    #pragma unroll
    for (int t = 0; t < 4; ++t) acc[t] = (v8f){0.f,0.f,0.f,0.f,0.f,0.f,0.f,0.f};

    for (int k0 = 0; k0 < DIM; k0 += 32) {
        // ---- A tile: 16 tokens x 32 K, fp32 -> bf16 hi/lo in WMMA layout.
        // Lane (m, kh) holds K = {kh*16+0..15} contiguous as bf16 pairs.
        v16bf ahi, alo;
        #pragma unroll
        for (int q = 0; q < 2; ++q) {              // q=0: K+0..7, q=1: K+8..15
            const float* p = arow + k0 + 16 * kh + q * 8;
            v4f x0 = *(const v4f*)(p);
            v4f x1 = *(const v4f*)(p + 4);
            #pragma unroll
            for (int j = 0; j < 4; ++j) {
                float xa = x0[j];
                __bf16 ha = (__bf16)xa;
                ahi[q * 8 + j]     = ha;
                alo[q * 8 + j]     = (__bf16)(xa - (float)ha);
                float xb = x1[j];
                __bf16 hb = (__bf16)xb;
                ahi[q * 8 + 4 + j] = hb;
                alo[q * 8 + 4 + j] = (__bf16)(xb - (float)hb);
            }
        }

        // ---- B tiles: 4 x (16 experts x 32 K), preconverted bf16.
        // Lane (n=m, kh) holds 16 contiguous K values -> one 32 B load each.
        #pragma unroll
        for (int t = 0; t < 4; ++t) {
            const size_t off = (size_t)(t * 16 + m) * DIM + k0 + 16 * kh;
            v16bf bhi = *(const v16bf*)(Whi + off);
            v16bf blo = *(const v16bf*)(Wlo + off);
            acc[t] = __builtin_amdgcn_wmma_f32_16x16x32_bf16(
                false, ahi, false, bhi, (short)0, acc[t], false, false);
            acc[t] = __builtin_amdgcn_wmma_f32_16x16x32_bf16(
                false, ahi, false, blo, (short)0, acc[t], false, false);
            acc[t] = __builtin_amdgcn_wmma_f32_16x16x32_bf16(
                false, alo, false, bhi, (short)0, acc[t], false, false);
        }
    }

    // ---- sigmoid(logits) -> LDS. C/D layout: VGPR r, lane: row = r + 8*kh,
    // col = 16*t + m.
    float* myrows = smem + wave * TOKENS_PER_WAVE * LDS_STRIDE;
    #pragma unroll
    for (int t = 0; t < 4; ++t) {
        #pragma unroll
        for (int r = 0; r < 8; ++r) {
            float lg = acc[t][r];
            float s  = 1.0f / (1.0f + __expf(-lg));
            myrows[(r + 8 * kh) * LDS_STRIDE + t * 16 + m] = s;
        }
    }
    __syncthreads();

    // ---- routing: lanes 0..15 each own one token (serial scans reproduce
    // jax.lax.top_k tie-breaking: strict '>' keeps the lowest index).
    if (ln < 16) {
        const float* sp = myrows + ln * LDS_STRIDE;   // 64 sigmoid scores

        float sfc[64];                                // scores + bias
        #pragma unroll
        for (int e = 0; e < 64; ++e) sfc[e] = sp[e] + bias[e];

        // group score = sum of top-2 biased scores within each group of 8
        float gsum[8];
        #pragma unroll
        for (int g = 0; g < 8; ++g) {
            float m1 = -INFINITY, m2 = -INFINITY;
            #pragma unroll
            for (int j = 0; j < 8; ++j) {
                float v = sfc[g * 8 + j];
                if (v > m1) { m2 = m1; m1 = v; }
                else if (v > m2) { m2 = v; }
            }
            gsum[g] = m1 + m2;
        }

        // pick TOPK_GROUP groups
        unsigned gsel = 0;
        #pragma unroll
        for (int it = 0; it < TOPK_GROUP; ++it) {
            float best = -INFINITY; int bg = 0;
            #pragma unroll
            for (int g = 0; g < 8; ++g) {
                if (!((gsel >> g) & 1u) && gsum[g] > best) { best = gsum[g]; bg = g; }
            }
            gsel |= 1u << bg;
        }

        // top-8 experts over masked biased scores (masked -> 0.0, like ref)
        unsigned long long used = 0ull;
        int   idxs[TOP_K];
        float wts[TOP_K];
        float wsum = 0.0f;
        #pragma unroll
        for (int it = 0; it < TOP_K; ++it) {
            float best = -INFINITY; int bi = 0;
            #pragma unroll
            for (int e = 0; e < 64; ++e) {
                float v = ((used >> e) & 1ull) ? -INFINITY
                          : (((gsel >> (e >> 3)) & 1u) ? sfc[e] : 0.0f);
                if (v > best) { best = v; bi = e; }
            }
            used |= 1ull << bi;
            idxs[it] = bi;
            float w = sp[bi];          // gather from UNBIASED sigmoid scores
            wts[it] = w;
            wsum += w;
        }

        const float scale = ROUTE_SCALE / (wsum + 1e-20f);
        const int token = tokenBase + ln;
        #pragma unroll
        for (int j = 0; j < TOP_K; ++j) {
            out[(size_t)token * TOP_K + j] = (float)idxs[j];
            out[(size_t)N_TOKENS * TOP_K + (size_t)token * TOP_K + j] = wts[j] * scale;
        }
    }
}

// ---------------------------------------------------------------------------
extern "C" void kernel_launch(void* const* d_in, const int* in_sizes, int n_in,
                              void* d_out, int out_size, void* d_ws, size_t ws_size,
                              hipStream_t stream) {
    (void)in_sizes; (void)n_in; (void)out_size; (void)ws_size;
    const float* hidden = (const float*)d_in[0];   // [32768*2048] f32
    const float* weight = (const float*)d_in[1];   // [64*2048] f32
    const float* bias   = (const float*)d_in[2];   // [64] f32
    float* out = (float*)d_out;

    __bf16* whi = (__bf16*)d_ws;                   // 64*2048 bf16 = 256 KB
    __bf16* wlo = whi + (size_t)N_EXPERTS * DIM;   // 64*2048 bf16 = 256 KB

    split_weights_bf16<<<(N_EXPERTS * DIM + 255) / 256, 256, 0, stream>>>(
        weight, whi, wlo);

    router_fused<<<N_TOKENS / BLOCK_TOKENS, 256, 0, stream>>>(
        hidden, whi, wlo, bias, out);
}
